// GATModel_85555748536564
// MI455X (gfx1250) — matrix operand order repaired and checked by reading, not compiled, not run on hardware
//
#include <hip/hip_runtime.h>
#include <hip/hip_bf16.h>

#define N_NODES  50000
#define N_EDGES  800000
#define ETOT     (N_EDGES + N_NODES)   // self loops appended
#define NHEAD    4
#define CHAN     32
#define HID      128
#define NLAYER   4
#define NGRAPH   16
#define NEG_SLOPE 0.2f
#define EPS_BN    1e-5f

typedef __bf16 bf16_t;
typedef __attribute__((ext_vector_type(16))) __bf16 v16bf;
typedef __attribute__((ext_vector_type(8)))  float  v8f;

union Frag { v16bf v; float4 q[2]; };

// ---------------------------------------------------------------- helpers
__device__ __forceinline__ int f2ord(float f) {          // order-preserving float->int
    int i = __float_as_int(f);
    return (i < 0) ? (i ^ 0x7fffffff) : i;
}
__device__ __forceinline__ float ord2f(int i) {
    return __int_as_float((i < 0) ? (i ^ 0x7fffffff) : i);
}
__device__ __forceinline__ void edge_sd(const int* __restrict__ ei, int e, int& s, int& d) {
    if (e < N_EDGES) { s = ei[e]; d = ei[N_EDGES + e]; }
    else             { s = d = e - N_EDGES; }            // self loop
}

// ------------------------------------------------- weight prep (fp32 -> bf16, transposed to [N][K])
// layout: [0,4096)  layer0  Wt0[n(128)][k(32)]  (k<5 valid, rest 0)
//         4096 + (l-1)*16384 : layer l Wt[n(128)][k(128)]
__global__ void prep_weights_kernel(const float* __restrict__ W0,
                                    const float* __restrict__ Ws,
                                    bf16_t* __restrict__ wbf) {
    int t = blockIdx.x * blockDim.x + threadIdx.x;
    if (t >= 53248) return;
    if (t < 4096) {
        int n = t >> 5, k = t & 31;
        wbf[t] = (k < 5) ? (bf16_t)W0[k * HID + n] : (bf16_t)0.0f;
    } else {
        int t2 = t - 4096;
        int l = t2 >> 14;
        int r = t2 & 16383;
        int n = r >> 7, k = r & 127;
        wbf[t] = (bf16_t)Ws[l * 16384 + k * HID + n];
    }
}

// ------------------------------------------------- activations fp32 -> bf16 (zero-padded K)
__global__ void cast_act_kernel(const float* __restrict__ src, bf16_t* __restrict__ dst,
                                int fin, int kshift, int total) {
    int t = blockIdx.x * blockDim.x + threadIdx.x;
    if (t >= total) return;
    int kpad = 1 << kshift;
    int n = t >> kshift, k = t & (kpad - 1);
    dst[t] = (k < fin) ? (bf16_t)src[n * fin + k] : (bf16_t)0.0f;
}

// ------------------------------------------------- WMMA GEMM: C[N,128] = A[N,Kpad] * Bt[128,Kpad]^T
// block = 256 threads = 8 waves, wave w owns column tile w; grid.x = N/16 = 3125 row tiles.
__global__ void wmma_gemm_kernel(const bf16_t* __restrict__ A,
                                 const bf16_t* __restrict__ Bt,
                                 float* __restrict__ C, int Kpad) {
    const int lane = threadIdx.x & 31;
    const int wave = threadIdx.x >> 5;     // 0..7  column tile
    const int hi   = lane >> 4;            // half-wave select
    const int l16  = lane & 15;
    const int mBase = blockIdx.x * 16;

    const bf16_t* arow = A  + (size_t)(mBase + l16) * Kpad;
    const bf16_t* brow = Bt + (size_t)(wave * 16 + l16) * Kpad;

    v8f acc = {};
    for (int k0 = 0; k0 < Kpad; k0 += 32) {
        Frag a, b;
        // A 16x32 bf16 frag: lanes 0-15 hold K[k0..k0+8)+K[k0+16..k0+24), lanes 16-31 the other halves
        a.q[0] = *(const float4*)(arow + k0 + hi * 8);
        a.q[1] = *(const float4*)(arow + k0 + 16 + hi * 8);
        // B 32x16 bf16 frag: lane = column, same K split as A: K[k0+hi*8..+8) + K[k0+16+hi*8..+8)
        const bf16_t* bp = brow + k0 + hi * 8;
        b.q[0] = *(const float4*)(bp);
        b.q[1] = *(const float4*)(bp + 16);
        acc = __builtin_amdgcn_wmma_f32_16x16x32_bf16(false, a.v, false, b.v,
                                                      (short)0, acc, false, false);
    }
    // C/D layout: VGPR j -> M = j + hi*8, N = l16
    float* crow = C + (size_t)(mBase + hi * 8) * HID + wave * 16 + l16;
#pragma unroll
    for (int j = 0; j < 8; ++j)
        crow[(size_t)j * HID] = acc[j];
}

// ------------------------------------------------- per-node attention logits
__global__ void alpha_kernel(const float* __restrict__ hw,
                             const float* __restrict__ att_s,   // pre-offset by l*H*C
                             const float* __restrict__ att_d,
                             float* __restrict__ asrc, float* __restrict__ adst) {
    int t = blockIdx.x * blockDim.x + threadIdx.x;
    if (t >= N_NODES * NHEAD) return;
    int n = t >> 2, h = t & 3;
    const float4* hp = (const float4*)(hw + (size_t)n * HID + h * CHAN);
    const float4* ap = (const float4*)(att_s + h * CHAN);
    const float4* bp = (const float4*)(att_d + h * CHAN);
    float ss = 0.f, sd = 0.f;
#pragma unroll
    for (int i = 0; i < 8; ++i) {
        float4 hv = hp[i], av = ap[i], bv = bp[i];
        ss += hv.x * av.x + hv.y * av.y + hv.z * av.z + hv.w * av.w;
        sd += hv.x * bv.x + hv.y * bv.y + hv.z * bv.z + hv.w * bv.w;
    }
    asrc[t] = ss;
    adst[t] = sd;
}

// ------------------------------------------------- per-layer scratch init
__global__ void init_layer_kernel(float* __restrict__ aggr, int* __restrict__ emax,
                                  float* __restrict__ denom,
                                  float* __restrict__ bnsum, float* __restrict__ bnsum2) {
    int t = blockIdx.x * blockDim.x + threadIdx.x;
    if (t < N_NODES * HID) aggr[t] = 0.f;
    if (t < N_NODES * NHEAD) { emax[t] = (int)0x80000000; denom[t] = 0.f; }
    if (t < HID) { bnsum[t] = 0.f; bnsum2[t] = 0.f; }
}

// ------------------------------------------------- pass 1: segment max (ordered-int atomicMax)
__global__ void edge_max_kernel(const int* __restrict__ ei,
                                const float* __restrict__ asrc, const float* __restrict__ adst,
                                int* __restrict__ emax) {
    int t = blockIdx.x * blockDim.x + threadIdx.x;
    if (t >= ETOT * NHEAD) return;
    int e = t >> 2, h = t & 3;
    int s, d; edge_sd(ei, e, s, d);
    float v = asrc[s * NHEAD + h] + adst[d * NHEAD + h];
    v = (v > 0.f) ? v : NEG_SLOPE * v;                    // leaky relu
    atomicMax(&emax[d * NHEAD + h], f2ord(v));
}

// ------------------------------------------------- pass 2: exp + denominator
__global__ void edge_exp_kernel(const int* __restrict__ ei,
                                const float* __restrict__ asrc, const float* __restrict__ adst,
                                const int* __restrict__ emax,
                                float* __restrict__ denom, float* __restrict__ wexp) {
    int t = blockIdx.x * blockDim.x + threadIdx.x;
    if (t >= ETOT * NHEAD) return;
    int e = t >> 2, h = t & 3;
    int s, d; edge_sd(ei, e, s, d);
    float v = asrc[s * NHEAD + h] + adst[d * NHEAD + h];
    v = (v > 0.f) ? v : NEG_SLOPE * v;
    float w = expf(v - ord2f(emax[d * NHEAD + h]));
    wexp[t] = w;
    atomicAdd(&denom[d * NHEAD + h], w);
}

// ------------------------------------------------- pass 3: weighted aggregation (L2-resident atomics)
// block = 256 threads = 2 edges x 128 channels; coalesced 512B row reads + contiguous atomic adds
__global__ void edge_aggr_kernel(const int* __restrict__ ei,
                                 const float* __restrict__ hw,
                                 const float* __restrict__ wexp,
                                 const float* __restrict__ denom,
                                 float* __restrict__ aggr) {
    int e = blockIdx.x * 2 + (threadIdx.x >> 7);
    if (e >= ETOT) return;
    int c = threadIdx.x & 127;
    int h = c >> 5;
    int s, d; edge_sd(ei, e, s, d);
    float alpha = wexp[e * NHEAD + h] / (denom[d * NHEAD + h] + 1e-16f);
    atomicAdd(&aggr[(size_t)d * HID + c], hw[(size_t)s * HID + c] * alpha);
}

// ------------------------------------------------- BN statistics (column sums, coalesced)
__global__ void bn_stats_kernel(const float* __restrict__ aggr,
                                float* __restrict__ bnsum, float* __restrict__ bnsum2) {
    int c = threadIdx.x;                                  // 0..127
    float s = 0.f, s2 = 0.f;
    for (int row = blockIdx.x; row < N_NODES; row += gridDim.x) {
        float v = aggr[(size_t)row * HID + c];
        s += v; s2 += v * v;
    }
    atomicAdd(&bnsum[c], s);
    atomicAdd(&bnsum2[c], s2);
}

// ------------------------------------------------- BN apply + SiLU (bias cancels inside BN exactly)
__global__ void bn_apply_kernel(const float* __restrict__ aggr,
                                const float* __restrict__ bnsum, const float* __restrict__ bnsum2,
                                const float* __restrict__ gamma, const float* __restrict__ beta,
                                float* __restrict__ xout) {
    int t = blockIdx.x * blockDim.x + threadIdx.x;
    if (t >= N_NODES * HID) return;
    int c = t & (HID - 1);
    const float invN = 1.0f / (float)N_NODES;
    float mu  = bnsum[c] * invN;
    float var = bnsum2[c] * invN - mu * mu;
    float rs  = rsqrtf(var + EPS_BN);
    float y = (aggr[t] - mu) * rs * gamma[c] + beta[c];
    xout[t] = y / (1.0f + expf(-y));                      // SiLU
}

// ------------------------------------------------- pooling
__global__ void pool_init_kernel(float* __restrict__ pool, float* __restrict__ cnt) {
    int t = blockIdx.x * blockDim.x + threadIdx.x;
    if (t < NGRAPH * HID) pool[t] = 0.f;
    if (t < NGRAPH) cnt[t] = 0.f;
}
__global__ void pool_kernel(const float* __restrict__ x, const int* __restrict__ batch,
                            float* __restrict__ pool, float* __restrict__ cnt) {
    int t = blockIdx.x * blockDim.x + threadIdx.x;
    if (t >= N_NODES * HID) return;
    int n = t >> 7, c = t & 127;
    int g = batch[n];
    atomicAdd(&pool[g * HID + c], x[t]);
    if (c == 0) atomicAdd(&cnt[g], 1.0f);
}

// ------------------------------------------------- final MLP: 128 -> 64 (SiLU) -> 1 (sigmoid)
__global__ void mlp_kernel(const float* __restrict__ pool, const float* __restrict__ cnt,
                           const float* __restrict__ fc1w, const float* __restrict__ fc1b,
                           const float* __restrict__ fc2w, const float* __restrict__ fc2b,
                           float* __restrict__ out) {
    __shared__ float sdata[64];
    int g = blockIdx.x, j = threadIdx.x;
    float cg = cnt[g]; cg = (cg > 1.0f) ? cg : 1.0f;
    float inv = 1.0f / cg;
    float z = fc1b[j];
    for (int k = 0; k < HID; ++k)
        z += pool[g * HID + k] * inv * fc1w[k * 64 + j];
    z = z / (1.0f + expf(-z));                            // SiLU
    sdata[j] = z * fc2w[j];
    __syncthreads();
    for (int s = 32; s > 0; s >>= 1) {
        if (j < s) sdata[j] += sdata[j + s];
        __syncthreads();
    }
    if (j == 0) {
        float y = sdata[0] + fc2b[0];
        out[g] = 1.0f / (1.0f + expf(-y));
    }
}

// ================================================= host side
extern "C" void kernel_launch(void* const* d_in, const int* in_sizes, int n_in,
                              void* d_out, int out_size, void* d_ws, size_t ws_size,
                              hipStream_t stream) {
    const float* d_x     = (const float*)d_in[0];   // [N,5]
    const int*   d_ei    = (const int*)  d_in[1];   // [2,E] flat
    const int*   d_batch = (const int*)  d_in[2];   // [N]
    const float* d_W0    = (const float*)d_in[4];   // [5,128]
    const float* d_Ws    = (const float*)d_in[5];   // [3,128,128]
    const float* d_as    = (const float*)d_in[6];   // [L,H,C]
    const float* d_ad    = (const float*)d_in[7];   // [L,H,C]
    // d_in[8] = bias (cancels exactly inside BN -> skipped)
    const float* d_gamma = (const float*)d_in[9];   // [L,128]
    const float* d_beta  = (const float*)d_in[10];  // [L,128]
    const float* d_fc1w  = (const float*)d_in[11];
    const float* d_fc1b  = (const float*)d_in[12];
    const float* d_fc2w  = (const float*)d_in[13];
    const float* d_fc2b  = (const float*)d_in[14];
    float* out = (float*)d_out;

    // ---- workspace layout (all recomputed every call -> deterministic)
    char* ws = (char*)d_ws;
    size_t off = 0;
    auto take = [&](size_t bytes) -> char* {
        char* p = ws + off;
        off = (off + bytes + 255) & ~(size_t)255;
        return p;
    };
    bf16_t* wbf   = (bf16_t*)take(53248 * sizeof(bf16_t));               // bf16 transposed weights
    bf16_t* hbf   = (bf16_t*)take((size_t)N_NODES * HID * sizeof(bf16_t));
    float*  xbuf  = (float*) take((size_t)N_NODES * HID * sizeof(float)); // activations (BN output)
    float*  hw    = (float*) take((size_t)N_NODES * HID * sizeof(float)); // GEMM output h = xW
    float*  aggr  = (float*) take((size_t)N_NODES * HID * sizeof(float)); // message accumulator
    float*  asrc  = (float*) take((size_t)N_NODES * NHEAD * sizeof(float));
    float*  adst  = (float*) take((size_t)N_NODES * NHEAD * sizeof(float));
    int*    emax  = (int*)   take((size_t)N_NODES * NHEAD * sizeof(int));
    float*  denom = (float*) take((size_t)N_NODES * NHEAD * sizeof(float));
    float*  wexp  = (float*) take((size_t)ETOT * NHEAD * sizeof(float));
    float*  bnsum = (float*) take(HID * sizeof(float));
    float*  bnsum2= (float*) take(HID * sizeof(float));
    float*  pool  = (float*) take(NGRAPH * HID * sizeof(float));
    float*  cnt   = (float*) take(NGRAPH * sizeof(float));
    (void)ws_size; (void)in_sizes; (void)n_in; (void)out_size;

    // ---- weights -> bf16 once per call
    prep_weights_kernel<<<(53248 + 255) / 256, 256, 0, stream>>>(d_W0, d_Ws, wbf);

    const int edgeBlocks = (ETOT * NHEAD + 255) / 256;
    const int nodeHidBlocks = (N_NODES * HID + 255) / 256;

    for (int l = 0; l < NLAYER; ++l) {
        const int fin    = (l == 0) ? 5 : HID;
        const int kshift = (l == 0) ? 5 : 7;       // Kpad = 32 or 128
        const int kpad   = 1 << kshift;
        const float*  src = (l == 0) ? d_x : xbuf;
        const bf16_t* Bt  = (l == 0) ? wbf : (wbf + 4096 + (size_t)(l - 1) * 16384);

        // 1) cast activations to bf16 (zero-padded K)
        int castTotal = N_NODES * kpad;
        cast_act_kernel<<<(castTotal + 255) / 256, 256, 0, stream>>>(src, hbf, fin, kshift, castTotal);

        // 2) WMMA GEMM: hw = hbf @ W  (3125 row tiles x 8 wave column tiles)
        wmma_gemm_kernel<<<N_NODES / 16, 256, 0, stream>>>(hbf, Bt, hw, kpad);

        // 3) attention logits per (node, head)
        alpha_kernel<<<(N_NODES * NHEAD + 255) / 256, 256, 0, stream>>>(
            hw, d_as + (size_t)l * NHEAD * CHAN, d_ad + (size_t)l * NHEAD * CHAN, asrc, adst);

        // 4) scratch init
        init_layer_kernel<<<nodeHidBlocks, 256, 0, stream>>>(aggr, emax, denom, bnsum, bnsum2);

        // 5) segment softmax (3 passes, L2-resident atomics)
        edge_max_kernel <<<edgeBlocks, 256, 0, stream>>>(d_ei, asrc, adst, emax);
        edge_exp_kernel <<<edgeBlocks, 256, 0, stream>>>(d_ei, asrc, adst, emax, denom, wexp);
        edge_aggr_kernel<<<(ETOT + 1) / 2, 256, 0, stream>>>(d_ei, hw, wexp, denom, aggr);

        // 6) batch norm + SiLU -> xbuf
        bn_stats_kernel<<<480, HID, 0, stream>>>(aggr, bnsum, bnsum2);
        bn_apply_kernel<<<nodeHidBlocks, 256, 0, stream>>>(
            aggr, bnsum, bnsum2, d_gamma + (size_t)l * HID, d_beta + (size_t)l * HID, xbuf);
    }

    // ---- mean pool + MLP head
    pool_init_kernel<<<(NGRAPH * HID + 255) / 256, 256, 0, stream>>>(pool, cnt);
    pool_kernel<<<nodeHidBlocks, 256, 0, stream>>>(xbuf, d_batch, pool, cnt);
    mlp_kernel<<<NGRAPH, 64, 0, stream>>>(pool, cnt, d_fc1w, d_fc1b, d_fc2w, d_fc2b, out);
}
